// SNNModel_6588479832546
// MI455X (gfx1250) — matrix operand order repaired and checked by reading, not compiled, not run on hardware
//
#include <hip/hip_runtime.h>

// ---------------------------------------------------------------------------
// SNN forward (snntorch-style Leaky, reset-by-subtract) for MI455X / gfx1250.
//
// All five big GEMMs run on v_wmma_f32_16x16x32_f16: spikes are exactly {0,1}
// in f16, weights ~U(-0.03,0.03) are safely f16. Weights (36 MB as f16) are
// L2-resident (192 MB) and reused 64x, so HBM is irrelevant; the design goal
// is minimizing L2 traffic and keeping the XDL pipes fed:
//  - waves of a block share the SAME B (weight) stream -> 8-way reuse via the
//    64 KB WGP$, ~3x less L2 traffic than sharing A.
//  - register double-buffered K loop: next tile's loads are in flight while
//    the current 4 WMMAs execute.
// ---------------------------------------------------------------------------

typedef _Float16 v16h __attribute__((ext_vector_type(16)));
typedef _Float16 v8h  __attribute__((ext_vector_type(8)));
typedef float    v8f  __attribute__((ext_vector_type(8)));

#define TSTEPS   64
#define BATCH    256
#define INSZ     1024
#define HIDDEN   2048
#define NLAYERS  4
#define OUTSZ    10
// BETA = 0.5f, THRESHOLD = 1.0f (hard-coded below)

// ---------------------------------------------------------------- converters
__global__ void __launch_bounds__(256)
f32_to_f16_kernel(const float* __restrict__ in, _Float16* __restrict__ out, int n) {
  for (int i = blockIdx.x * blockDim.x + threadIdx.x; i < n;
       i += gridDim.x * blockDim.x)
    out[i] = (_Float16)in[i];
}

__global__ void __launch_bounds__(256)
zero_f32_kernel(float* __restrict__ p, int n) {
  for (int i = blockIdx.x * blockDim.x + threadIdx.x; i < n;
       i += gridDim.x * blockDim.x)
    p[i] = 0.0f;
}

// ---- fragment loaders (wave32 16-bit WMMA, contiguous k = 16*half + s) --------------------
// A 16x32 (MxK): lane(l16)=M; lanes 0-15 hold K 0-15, lanes 16-31 hold
// K 16-31 (same k order as B). `p` already includes the half*16 shift.
__device__ __forceinline__ v16h load_a_frag(const _Float16* p) {
  return *(const v16h*)(p);
}
// B 32x16 (KxN): lane(l16)=N; lanes 0-15 hold K 0-15, lanes 16-31 K 16-31.
// W is [N,K] row-major -> the 16 f16 are contiguous (32B, 32B-aligned).
__device__ __forceinline__ v16h load_b_frag(const _Float16* p) {
  return *(const v16h*)(p);
}

// ------------------------------------------------- fused GEMM + Leaky update
// C[m,n] = sum_k A[m,k] * W[n,k] + bias[n], then LIF update on mem/spk.
// One wave owns a 16(M) x 64(N) tile. Mapping: the 8 waves of a block share
// one ntile-group (same B stream -> WGP$ reuse) and take 8 distinct mtiles.
// Grid is exact (512 waves), so EXEC is all-ones for WMMA as the ISA requires.
__global__ void __launch_bounds__(256)
gemm_leaky_kernel(const _Float16* __restrict__ A,
                  const _Float16* __restrict__ W,
                  const float*    __restrict__ bias,
                  float*          __restrict__ mem,
                  _Float16*       __restrict__ spk,
                  int K)
{
  const int tid  = blockIdx.x * blockDim.x + threadIdx.x;
  const int wid  = tid >> 5;            // global wave id, 0..511
  const int lane = threadIdx.x & 31;
  const int half = lane >> 4;           // 0: lanes 0-15, 1: lanes 16-31
  const int l16  = lane & 15;

  const int ngrp  = wid >> 4;           // 0..31 (HIDDEN/64): 2 blocks/ngrp
  const int mtile = wid & 15;           // 0..15 (BATCH/16): 8 mtiles/block
  const int mbase = mtile * 16;
  const int nbase = ngrp * 64;

  v8f c[4];
#pragma unroll
  for (int j = 0; j < 4; ++j)
#pragma unroll
    for (int r = 0; r < 8; ++r) c[j][r] = 0.0f;

  const _Float16* arow  = A + (size_t)(mbase + l16) * K + half * 16;
  const _Float16* brow0 = W + (size_t)(nbase + l16) * K + half * 16;
  const size_t bstride = (size_t)16 * K;  // 16 W-rows per B tile

  // ---- software-pipelined K loop, register double-buffered (K%64==0) ----
  v16h a0 = load_a_frag(arow);
  v16h b0[4];
#pragma unroll
  for (int j = 0; j < 4; ++j) b0[j] = load_b_frag(brow0 + j * bstride);

  for (int kt = 0; kt < K; kt += 64) {
    if (kt + 128 < K) {                      // keep the B stream warm
      __builtin_prefetch(arow + kt + 128, 0, 1);   // global_prefetch_b8
      __builtin_prefetch(brow0 + kt + 128, 0, 1);
    }
    // issue loads for kt+32 while WMMAs of kt execute
    v16h a1 = load_a_frag(arow + kt + 32);
    v16h b1[4];
#pragma unroll
    for (int j = 0; j < 4; ++j) b1[j] = load_b_frag(brow0 + j * bstride + kt + 32);

#pragma unroll
    for (int j = 0; j < 4; ++j)
      c[j] = __builtin_amdgcn_wmma_f32_16x16x32_f16(false, a0, false, b0[j],
                                                    (short)0, c[j], false, false);

    if (kt + 64 < K) {                       // uniform branch: EXEC untouched
      a0 = load_a_frag(arow + kt + 64);
#pragma unroll
      for (int j = 0; j < 4; ++j) b0[j] = load_b_frag(brow0 + j * bstride + kt + 64);
    }

#pragma unroll
    for (int j = 0; j < 4; ++j)
      c[j] = __builtin_amdgcn_wmma_f32_16x16x32_f16(false, a1, false, b1[j],
                                                    (short)0, c[j], false, false);
  }

  // Fused snn.Leaky (reset_mechanism='subtract') epilogue.
  // C/D layout: VGPR r -> row M = r + 8*half; lane -> col N = l16.
#pragma unroll
  for (int j = 0; j < 4; ++j) {
    const int n = nbase + j * 16 + l16;
    const float bn = bias[n];
#pragma unroll
    for (int r = 0; r < 8; ++r) {
      const int m = mbase + r + 8 * half;
      const size_t idx = (size_t)m * HIDDEN + n;
      const float cur   = c[j][r] + bn;
      const float mv    = mem[idx];
      const float reset = (mv - 1.0f > 0.0f) ? 1.0f : 0.0f;  // from OLD mem
      const float nm    = 0.5f * mv + cur - reset;           // beta*mem+cur-reset*thr
      mem[idx] = nm;
      spk[idx] = (nm - 1.0f > 0.0f) ? (_Float16)1.0f : (_Float16)0.0f;
    }
  }
}

// ------------------------------------------------- output head (N=10) + LIF
// One wave per (batch, output) pair; shuffle reduction over K=2048.
// A 16x16 WMMA tile would be 37.5% wasted on N=10, so VALU is the right tool.
__global__ void __launch_bounds__(320)
head_leaky_kernel(const _Float16* __restrict__ spk,   // [BATCH, HIDDEN]
                  const float*    __restrict__ Wl,    // [OUTSZ, HIDDEN]
                  const float*    __restrict__ bl,    // [OUTSZ]
                  float*          __restrict__ meml,  // [BATCH, OUTSZ]
                  float*          __restrict__ out_spk,
                  float*          __restrict__ out_mem)
{
  const int b    = blockIdx.x;
  const int o    = threadIdx.x >> 5;    // 0..9
  const int lane = threadIdx.x & 31;
  const _Float16* srow = spk + (size_t)b * HIDDEN;
  const float*    wrow = Wl  + (size_t)o * HIDDEN;

  float s = 0.0f;
  for (int k = lane; k < HIDDEN; k += 32)
    s += (float)srow[k] * wrow[k];
#pragma unroll
  for (int off = 16; off > 0; off >>= 1)
    s += __shfl_xor(s, off, 32);

  if (lane == 0) {
    const int idx = b * OUTSZ + o;
    const float cur   = s + bl[o];
    const float mv    = meml[idx];
    const float reset = (mv - 1.0f > 0.0f) ? 1.0f : 0.0f;
    const float nm    = 0.5f * mv + cur - reset;
    meml[idx]    = nm;
    out_mem[idx] = nm;
    out_spk[idx] = (nm - 1.0f > 0.0f) ? 1.0f : 0.0f;
  }
}

// ---------------------------------------------------------------------------
extern "C" void kernel_launch(void* const* d_in, const int* in_sizes, int n_in,
                              void* d_out, int out_size, void* d_ws, size_t ws_size,
                              hipStream_t stream) {
  (void)in_sizes; (void)n_in; (void)out_size; (void)ws_size;

  const float* x  = (const float*)d_in[0];  // [64,256,1024]
  const float* Wf = (const float*)d_in[1];  // [2048,1024]
  const float* bf = (const float*)d_in[2];  // [2048]
  const float* Ws = (const float*)d_in[3];  // [4,2048,2048]
  const float* bs = (const float*)d_in[4];  // [4,2048]
  const float* Wl = (const float*)d_in[5];  // [10,2048]
  const float* bl = (const float*)d_in[6];  // [10]
  float* out = (float*)d_out;               // spk_rec[64,256,10] ++ mem_rec[64,256,10]

  // ---- workspace carving (256B aligned sections) ----
  char* ws = (char*)d_ws;
  size_t off = 0;
  auto carve = [&](size_t bytes) {
    char* p = ws + off;
    off = (off + bytes + 255) & ~(size_t)255;
    return p;
  };
  const size_t xN  = (size_t)TSTEPS * BATCH * INSZ;
  const size_t WfN = (size_t)HIDDEN * INSZ;
  const size_t WsN = (size_t)NLAYERS * HIDDEN * HIDDEN;

  _Float16* xh   = (_Float16*)carve(xN  * 2);                       // 32 MB
  _Float16* Wfh  = (_Float16*)carve(WfN * 2);                       //  4 MB
  _Float16* Wsh  = (_Float16*)carve(WsN * 2);                       // 32 MB
  _Float16* spk0 = (_Float16*)carve((size_t)BATCH * HIDDEN * 2);    //  1 MB
  _Float16* spk1 = (_Float16*)carve((size_t)BATCH * HIDDEN * 2);    //  1 MB
  float*    memf = (float*)carve((size_t)BATCH * HIDDEN * 4);       //  2 MB
  float*    mems = (float*)carve((size_t)NLAYERS * BATCH * HIDDEN * 4); // 8 MB
  float*    meml = (float*)carve((size_t)BATCH * OUTSZ * 4);

  // ---- one-time precision conversion + state init (re-run per call: deterministic)
  f32_to_f16_kernel<<<2048, 256, 0, stream>>>(x,  xh,  (int)xN);
  f32_to_f16_kernel<<<2048, 256, 0, stream>>>(Wf, Wfh, (int)WfN);
  f32_to_f16_kernel<<<2048, 256, 0, stream>>>(Ws, Wsh, (int)WsN);
  zero_f32_kernel<<<512, 256, 0, stream>>>(memf, BATCH * HIDDEN);
  zero_f32_kernel<<<512, 256, 0, stream>>>(mems, NLAYERS * BATCH * HIDDEN);
  zero_f32_kernel<<<16,  256, 0, stream>>>(meml, BATCH * OUTSZ);

  // ---- 64 sequential time steps; stream ordering carries the dependences
  for (int t = 0; t < TSTEPS; ++t) {
    // feed-forward layer: [256,1024] x [2048,1024]^T
    gemm_leaky_kernel<<<64, 256, 0, stream>>>(
        xh + (size_t)t * BATCH * INSZ, Wfh, bf, memf, spk0, INSZ);

    _Float16* sin = spk0;
    _Float16* sout = spk1;
    for (int i = 0; i < NLAYERS; ++i) {
      gemm_leaky_kernel<<<64, 256, 0, stream>>>(
          sin, Wsh + (size_t)i * HIDDEN * HIDDEN, bs + i * HIDDEN,
          mems + (size_t)i * BATCH * HIDDEN, sout, HIDDEN);
      _Float16* tmp = sin; sin = sout; sout = tmp;
    }
    // head: [256,2048] x [10,2048]^T + LIF; writes spk_rec[t], mem_rec[t]
    head_leaky_kernel<<<BATCH, 320, 0, stream>>>(
        sin, Wl, bl, meml,
        out + (size_t)t * BATCH * OUTSZ,
        out + (size_t)TSTEPS * BATCH * OUTSZ + (size_t)t * BATCH * OUTSZ);
  }
}
